// CrossGraphEncoder_79173427135043
// MI455X (gfx1250) — matrix operand order repaired and checked by reading, not compiled, not run on hardware
//
#include <hip/hip_runtime.h>

#define B_  32
#define NA_ 200
#define NG_ 1000
#define M_  1200
#define D_  128
#define K_  8
#define L_  4
#define GS_ 10

#define TOTAL_NODES (B_ * M_)          // 38400
#define NODES_PER_BLOCK 8              // 4 waves * 2 nodes
#define LAYER_BLOCKS (TOTAL_NODES / NODES_PER_BLOCK)   // 4800
#define WSTG_ELEMS 16384u              // one 32KB weight stage (bf16 elems)

typedef __attribute__((ext_vector_type(16))) __bf16        v16bf;
typedef __attribute__((ext_vector_type(8)))  float         v8f;
typedef __attribute__((ext_vector_type(4)))  unsigned int  tdm_v4u;
typedef __attribute__((ext_vector_type(8)))  int           tdm_v8i;
typedef __attribute__((ext_vector_type(4)))  int           tdm_v4i;

union FragU {
  v16bf         v;
  unsigned int  u[8];
  unsigned short h[16];
};

__device__ __forceinline__ unsigned short f2bf(float f) {
  unsigned int u = __float_as_uint(f);
  unsigned int r = u + 0x7FFFu + ((u >> 16) & 1u);   // round-to-nearest-even
  return (unsigned short)(r >> 16);
}

// ---------------------------------------------------------------------------
// A-fragment (16x32 bf16, row = edge) from a row pointer (global or LDS).
// 16-bit A layout: lanes 0-15 / 16-31 both hold row (lane&15);
// VGPR v holds K = kt*32 + (v<4 ? 2v : 16+2(v-4)) + (lane>>4)*8, pairs packed.
// ---------------------------------------------------------------------------
__device__ __forceinline__ v16bf load_a16(const unsigned short* row, int cbase, int hi) {
  FragU f;
#pragma unroll
  for (int v = 0; v < 8; ++v) {
    int kl = ((v < 4) ? (v * 2) : (16 + (v - 4) * 2)) + hi * 8;
    f.u[v] = *(const unsigned int*)(row + cbase + kl);
  }
  return f.v;
}

// B-fragment: pre-swizzled blocks of 512 bf16; lane reads 32 contiguous bytes.
__device__ __forceinline__ v16bf load_b16(const unsigned int* wbufU, int blk, int lane) {
  FragU f;
  const unsigned int* p = wbufU + blk * 256 + lane * 8;
#pragma unroll
  for (int j = 0; j < 8; ++j) f.u[j] = p[j];
  return f.v;
}

// ---------------------------------------------------------------------------
// Weight staging: TDM descriptor copy of 16384 bf16 (32KB) global -> LDS.
// ---------------------------------------------------------------------------
#if defined(__has_builtin)
#if __has_builtin(__builtin_amdgcn_tensor_load_to_lds) && __has_builtin(__builtin_amdgcn_s_wait_tensorcnt)
#define USE_TDM 1
#endif
#endif
#ifndef USE_TDM
#define USE_TDM 0
#endif

__device__ __forceinline__ void stage_weights(const unsigned short* __restrict__ gsrc,
                                              unsigned int* ldsU, int tid) {
#if USE_TDM
  if (tid < 32) {  // wave 0 issues the tensor DMA; EXEC is ignored by TDM
    unsigned long long ga = (unsigned long long)(size_t)gsrc;
    unsigned int lo = (unsigned int)(unsigned long long)(size_t)ldsU;  // LDS byte offset
    tdm_v4u g0;
    g0[0] = 1u;                                                  // count=1, user desc
    g0[1] = lo;                                                  // lds_addr
    g0[2] = (unsigned int)ga;                                    // global_addr[31:0]
    g0[3] = (unsigned int)((ga >> 32) & 0x01FFFFFFull) | (2u << 30);  // addr hi | type=2
    tdm_v8i g1;
    g1[0] = (int)(1u << 16);                                     // data_size=1 (2B), mask=0
    g1[1] = (int)((WSTG_ELEMS & 0xFFFFu) << 16);                 // tensor_dim0 lo16
    g1[2] = (int)(((WSTG_ELEMS >> 16) & 0xFFFFu) | (1u << 16));  // dim0 hi | tensor_dim1=1
    g1[3] = (int)((WSTG_ELEMS & 0xFFFFu) << 16);                 // tile_dim0
    g1[4] = 0;                                                   // tile_dim1/2 unused
    g1[5] = (int)WSTG_ELEMS;                                     // dim0 stride lo32
    g1[6] = 0;
    g1[7] = 0;
    tdm_v4i z4 = {0, 0, 0, 0};
#if __clang_major__ >= 23
    tdm_v8i z8 = {0, 0, 0, 0, 0, 0, 0, 0};
    __builtin_amdgcn_tensor_load_to_lds(g0, g1, z4, z4, z8, 0);
#else
    __builtin_amdgcn_tensor_load_to_lds(g0, g1, z4, z4, 0);
#endif
    __builtin_amdgcn_s_wait_tensorcnt(0);
  }
  __syncthreads();
#else
  const unsigned int* g32 = (const unsigned int*)gsrc;
  for (int i = tid; i < (int)(WSTG_ELEMS / 2); i += 128) ldsU[i] = g32[i];
  __syncthreads();
#endif
}

// ---------------------------------------------------------------------------
// Prep: convert w1/w2 fp32 -> bf16 in WMMA B-fragment-major layout.
// swz1 blocks: (l, kt:8, nt:8) of [lane][i]; swz2 blocks: (l, kt:4, nt:8).
// ---------------------------------------------------------------------------
__global__ void prep_weights_kernel(const float* __restrict__ w1, const float* __restrict__ w2,
                                    unsigned short* __restrict__ swz1,
                                    unsigned short* __restrict__ swz2) {
  int id = blockIdx.x * blockDim.x + threadIdx.x;
  const int T1 = L_ * 64 * 512;
  const int T2 = L_ * 32 * 512;
  if (id < T1) {
    int i = id & 15, ln = (id >> 4) & 31, blk = id >> 9;
    int nt = blk & 7, kt = (blk >> 3) & 7, l = blk >> 6;
    int k = kt * 32 + (ln >> 4) * 8 + (i < 8 ? i : 8 + i);
    int col = nt * 16 + (ln & 15);
    swz1[id] = f2bf(w1[((size_t)l * 257 + k) * 128 + col]);
  }
  if (id < T2) {
    int i = id & 15, ln = (id >> 4) & 31, blk = id >> 9;
    int nt = blk & 7, kt = (blk >> 3) & 3, l = blk >> 5;
    int k = kt * 32 + (ln >> 4) * 8 + (i < 8 ? i : 8 + i);
    int col = nt * 16 + (ln & 15);
    swz2[id] = f2bf(w2[((size_t)l * 128 + k) * 128 + col]);
  }
}

// ---------------------------------------------------------------------------
// Node features (fp32 master + bf16 shadow) and positions.
// ---------------------------------------------------------------------------
__global__ void build_x_kernel(const int* __restrict__ atom_types,
                               const float* __restrict__ grid_codes,
                               const float* __restrict__ proj_w,
                               const float* __restrict__ proj_b,
                               float* __restrict__ xf, unsigned short* __restrict__ xb) {
  const long total = (long)B_ * M_ * D_;
  for (long id = (long)blockIdx.x * blockDim.x + threadIdx.x; id < total;
       id += (long)gridDim.x * blockDim.x) {
    int c = (int)(id % D_);
    long n = id / D_;
    int m = (int)(n % M_);
    int b = (int)(n / M_);
    float v;
    if (m < NA_) v = proj_w[atom_types[b * NA_ + m] * D_ + c] + proj_b[c];
    else         v = grid_codes[(size_t)(m - NA_) * D_ + c];
    xf[id] = v;
    xb[id] = f2bf(v);
  }
}

__global__ void build_pos_kernel(const float* __restrict__ atom_coords, float* __restrict__ pos) {
  const int total = B_ * M_ * 3;
  for (int id = blockIdx.x * blockDim.x + threadIdx.x; id < total;
       id += gridDim.x * blockDim.x) {
    int a = id % 3;
    int n = id / 3;
    int m = n % M_;
    int b = n / M_;
    float v;
    if (m < NA_) {
      v = atom_coords[(b * NA_ + m) * 3 + a];
    } else {
      int g = m - NA_;
      int idx = (a == 0) ? (g / (GS_ * GS_)) : (a == 1) ? ((g / GS_) % GS_) : (g % GS_);
      v = -1.0f + (float)idx * (2.0f / (float)(GS_ - 1));  // linspace(-1,1,GS)
    }
    pos[id] = v;
  }
}

// ---------------------------------------------------------------------------
// KNN: one workgroup per graph; positions in LDS; top-8 insertion sort with
// (d2, idx) lexicographic order matching jax.lax.top_k tie-breaking.
// ---------------------------------------------------------------------------
__global__ __launch_bounds__(256) void knn_kernel(const float* __restrict__ pos,
                                                  int* __restrict__ nbr,
                                                  float* __restrict__ dist) {
  __shared__ float px[M_], py[M_], pz[M_];
  int b = blockIdx.x;
  const float* pb = pos + (size_t)b * M_ * 3;
  for (int i = threadIdx.x; i < M_; i += blockDim.x) {
    px[i] = pb[i * 3 + 0];
    py[i] = pb[i * 3 + 1];
    pz[i] = pb[i * 3 + 2];
  }
  __syncthreads();
  for (int m = threadIdx.x; m < M_; m += blockDim.x) {
    float bd[K_];
    int   bi[K_];
#pragma unroll
    for (int s = 0; s < K_; ++s) { bd[s] = 3.4e38f; bi[s] = 0x7fffffff; }
    float x = px[m], y = py[m], z = pz[m];
    for (int j = 0; j < M_; ++j) {
      if (j == m) continue;  // diagonal gets +1e9 in ref -> never selected
      float dx = px[j] - x, dy = py[j] - y, dz = pz[j] - z;
      float d = __fadd_rn(__fadd_rn(__fmul_rn(dx, dx), __fmul_rn(dy, dy)), __fmul_rn(dz, dz));
      int id = j;
      if (d > bd[K_ - 1] || (d == bd[K_ - 1] && id > bi[K_ - 1])) continue;
#pragma unroll
      for (int s = 0; s < K_; ++s) {
        bool better = (d < bd[s]) || (d == bd[s] && id < bi[s]);
        float td = bd[s]; int ti = bi[s];
        if (better) { bd[s] = d; bi[s] = id; d = td; id = ti; }
      }
    }
#pragma unroll
    for (int s = 0; s < K_; ++s) {
      nbr[((size_t)b * M_ + m) * K_ + s]  = bi[s];
      dist[((size_t)b * M_ + m) * K_ + s] = sqrtf(bd[s]);
    }
  }
}

// ---------------------------------------------------------------------------
// Layer: fused edge-MLP (bf16 WMMA) + mean-over-K + residual + LayerNorm.
// 128 threads = 4 waves; wave owns 16 edges (2 nodes x 8 neighbors).
// ---------------------------------------------------------------------------
__global__ __launch_bounds__(128) void layer_kernel(
    const float* __restrict__ xf_in, const unsigned short* __restrict__ xb_in,
    float* __restrict__ xf_out, unsigned short* __restrict__ xb_out,
    const int* __restrict__ nbr, const float* __restrict__ dist,
    const unsigned short* __restrict__ swz1_l, const unsigned short* __restrict__ swz2_l,
    const float* __restrict__ w1_dist_row, const float* __restrict__ b1_l,
    const float* __restrict__ b2_l, const float* __restrict__ lns_l,
    const float* __restrict__ lnb_l) {
  __shared__ unsigned int  wbufU[WSTG_ELEMS / 2];          // 32 KB weight stage
  __shared__ unsigned short hS[4][16][D_];                 // 16 KB bf16 h staging
  __shared__ float dS[4][16];                              // edge distances
  __shared__ float yS[4][2][D_];                           // pre-LN activations
  __shared__ float statS[4][2][2];                         // mu, rsigma

  const int tid  = threadIdx.x;
  const int wave = tid >> 5;
  const int lane = tid & 31;
  const int hi   = lane >> 4;     // lane half
  const int r16  = lane & 15;     // edge row within the 16-edge tile

  const int nodePair = blockIdx.x * NODES_PER_BLOCK + wave * 2;  // global node id
  const int b = (blockIdx.x * NODES_PER_BLOCK) / M_;             // uniform per block

  // Per-lane edge bookkeeping (edge e == r16).
  const int gn  = nodePair + (r16 >> 3);       // global node of this edge
  const int kk  = r16 & 7;
  const int src = nbr[(size_t)gn * K_ + kk];   // in-graph neighbor index
  const unsigned short* rowN = xb_in + ((size_t)b * M_ + src) * D_;  // x_nbr row
  const unsigned short* rowS = xb_in + (size_t)gn * D_;              // x_self row
  if (hi == 0) dS[wave][r16] = dist[(size_t)gn * K_ + kk];
  __syncthreads();

  // ---- Stage 1: h = relu(m_in @ W1 + b1), m_in = [x_nbr | x_self | dist]
  v8f acc[8];
#pragma unroll
  for (int nt = 0; nt < 8; ++nt) {
    float bb = b1_l[nt * 16 + r16];
#pragma unroll
    for (int r = 0; r < 8; ++r) acc[nt][r] = bb;
  }

  stage_weights(swz1_l, wbufU, tid);               // W1 rows 0..127 (neighbor part)
  for (int kt = 0; kt < 4; ++kt) {
    v16bf a = load_a16(rowN, kt * 32, hi);
#pragma unroll
    for (int nt = 0; nt < 8; ++nt) {
      v16bf bf = load_b16(wbufU, kt * 8 + nt, lane);
      acc[nt] = __builtin_amdgcn_wmma_f32_16x16x32_bf16(false, a, false, bf,
                                                        (short)0, acc[nt], false, false);
    }
  }
  __syncthreads();
  stage_weights(swz1_l + WSTG_ELEMS, wbufU, tid);  // W1 rows 128..255 (self part)
  for (int kt = 0; kt < 4; ++kt) {
    v16bf a = load_a16(rowS, kt * 32, hi);
#pragma unroll
    for (int nt = 0; nt < 8; ++nt) {
      v16bf bf = load_b16(wbufU, kt * 8 + nt, lane);
      acc[nt] = __builtin_amdgcn_wmma_f32_16x16x32_bf16(false, a, false, bf,
                                                        (short)0, acc[nt], false, false);
    }
  }

  // dist rank-1 term (W1 row 256) + ReLU, then stage h as bf16 for stage 2.
#pragma unroll
  for (int nt = 0; nt < 8; ++nt) {
    float wd = w1_dist_row[nt * 16 + r16];
#pragma unroll
    for (int r = 0; r < 8; ++r) {
      float v = acc[nt][r] + dS[wave][r + 8 * hi] * wd;
      v = fmaxf(v, 0.0f);
      hS[wave][r + 8 * hi][nt * 16 + r16] = f2bf(v);
    }
  }
  __syncthreads();

  // ---- Stage 2: msg = h @ W2 + b2
  stage_weights(swz2_l, wbufU, tid);
  v8f acc2[8];
#pragma unroll
  for (int nt = 0; nt < 8; ++nt) {
    float bb = b2_l[nt * 16 + r16];
#pragma unroll
    for (int r = 0; r < 8; ++r) acc2[nt][r] = bb;
  }
  const unsigned short* hrow = &hS[wave][r16][0];
  for (int kt = 0; kt < 4; ++kt) {
    v16bf a = load_a16(hrow, kt * 32, hi);
#pragma unroll
    for (int nt = 0; nt < 8; ++nt) {
      v16bf bf = load_b16(wbufU, kt * 8 + nt, lane);
      acc2[nt] = __builtin_amdgcn_wmma_f32_16x16x32_bf16(false, a, false, bf,
                                                         (short)0, acc2[nt], false, false);
    }
  }

  // ---- Mean over K (8 C-rows per lane-half == 8 edges of one node) + residual
  const int myNode = nodePair + hi;
#pragma unroll
  for (int nt = 0; nt < 8; ++nt) {
    float s = 0.0f;
#pragma unroll
    for (int r = 0; r < 8; ++r) s += acc2[nt][r];
    s *= 0.125f;
    int col = nt * 16 + r16;
    yS[wave][hi][col] = xf_in[(size_t)myNode * D_ + col] + s;
  }

  // ---- LayerNorm over D (two-pass, matching jnp.var)
  if (r16 == 0) {
    float su = 0.0f;
    for (int c = 0; c < D_; ++c) su += yS[wave][hi][c];
    float mu = su * (1.0f / D_);
    float q = 0.0f;
    for (int c = 0; c < D_; ++c) { float t = yS[wave][hi][c] - mu; q += t * t; }
    statS[wave][hi][0] = mu;
    statS[wave][hi][1] = rsqrtf(q * (1.0f / D_) + 1e-5f);
  }
  float mu = statS[wave][hi][0];
  float rs = statS[wave][hi][1];
#pragma unroll
  for (int nt = 0; nt < 8; ++nt) {
    int col = nt * 16 + r16;
    float o = (yS[wave][hi][col] - mu) * rs * lns_l[col] + lnb_l[col];
    size_t oi = (size_t)myNode * D_ + col;
    xf_out[oi] = o;
    xb_out[oi] = f2bf(o);
  }
}

// ---------------------------------------------------------------------------
// Final slice: x[:, NA:, :] -> d_out
// ---------------------------------------------------------------------------
__global__ void copy_out_kernel(const float* __restrict__ xf, float* __restrict__ out) {
  const long total = (long)B_ * NG_ * D_;
  for (long id = (long)blockIdx.x * blockDim.x + threadIdx.x; id < total;
       id += (long)gridDim.x * blockDim.x) {
    int c = (int)(id % D_);
    long n = id / D_;
    int g = (int)(n % NG_);
    int b = (int)(n / NG_);
    out[id] = xf[((size_t)b * M_ + NA_ + g) * D_ + c];
  }
}

// ---------------------------------------------------------------------------
extern "C" void kernel_launch(void* const* d_in, const int* in_sizes, int n_in,
                              void* d_out, int out_size, void* d_ws, size_t ws_size,
                              hipStream_t stream) {
  (void)in_sizes; (void)n_in; (void)out_size; (void)ws_size;
  const float* atom_coords = (const float*)d_in[0];
  const int*   atom_types  = (const int*)d_in[1];
  const float* grid_codes  = (const float*)d_in[2];
  const float* proj_w      = (const float*)d_in[3];
  const float* proj_b      = (const float*)d_in[4];
  const float* w1          = (const float*)d_in[5];
  const float* b1          = (const float*)d_in[6];
  const float* w2          = (const float*)d_in[7];
  const float* b2          = (const float*)d_in[8];
  const float* lns         = (const float*)d_in[9];
  const float* lnb         = (const float*)d_in[10];
  float* out = (float*)d_out;

  // Workspace carve-up (all 256B aligned).
  char* ws = (char*)d_ws;
  size_t off = 0;
  auto carve = [&](size_t bytes) {
    void* p = ws + off;
    off = (off + bytes + 255) & ~(size_t)255;
    return p;
  };
  const size_t XN = (size_t)B_ * M_ * D_;
  float*          xf0  = (float*)carve(XN * 4);
  float*          xf1  = (float*)carve(XN * 4);
  unsigned short* xb0  = (unsigned short*)carve(XN * 2);
  unsigned short* xb1  = (unsigned short*)carve(XN * 2);
  float*          pos  = (float*)carve((size_t)B_ * M_ * 3 * 4);
  int*            nbr  = (int*)carve((size_t)B_ * M_ * K_ * 4);
  float*          dst  = (float*)carve((size_t)B_ * M_ * K_ * 4);
  unsigned short* swz1 = (unsigned short*)carve((size_t)L_ * 64 * 512 * 2);
  unsigned short* swz2 = (unsigned short*)carve((size_t)L_ * 32 * 512 * 2);

  prep_weights_kernel<<<512, 256, 0, stream>>>(w1, w2, swz1, swz2);
  build_x_kernel<<<4800, 256, 0, stream>>>(atom_types, grid_codes, proj_w, proj_b, xf0, xb0);
  build_pos_kernel<<<450, 256, 0, stream>>>(atom_coords, pos);
  knn_kernel<<<B_, 256, 0, stream>>>(pos, nbr, dst);

  for (int l = 0; l < L_; ++l) {
    const float*          xin  = (l & 1) ? xf1 : xf0;
    const unsigned short* xbin = (l & 1) ? xb1 : xb0;
    float*                xout = (l & 1) ? xf0 : xf1;
    unsigned short*       xbout= (l & 1) ? xb0 : xb1;
    layer_kernel<<<LAYER_BLOCKS, 128, 0, stream>>>(
        xin, xbin, xout, xbout, nbr, dst,
        swz1 + (size_t)l * 64 * 512, swz2 + (size_t)l * 32 * 512,
        w1 + ((size_t)l * 257 + 256) * 128,
        b1 + (size_t)l * 128, b2 + (size_t)l * 128,
        lns + (size_t)l * 128, lnb + (size_t)l * 128);
  }
  // L_=4 -> final activations land back in xf0.
  copy_out_kernel<<<4000, 256, 0, stream>>>(xf0, out);
}
